// MultiHeadAttention_274877906955
// MI455X (gfx1250) — matrix operand order repaired and checked by reading, not compiled, not run on hardware
//
#include <hip/hip_runtime.h>

typedef __bf16 bf16_t;
typedef __attribute__((ext_vector_type(8)))  bf16_t v8bf;
typedef __attribute__((ext_vector_type(16))) bf16_t v16bf;
typedef __attribute__((ext_vector_type(8)))  float  v8f;
typedef __attribute__((ext_vector_type(4)))  int    v4i;

#define BB  2
#define SS  2048
#define DD  1024
#define HH  16
#define HDD 64
#define BSZ (BB * SS)   // 4096 rows

// ---------- CDNA5 async global->LDS staging (ASYNCcnt path) ----------
#if __has_builtin(__builtin_amdgcn_global_load_async_to_lds_b128)
#define HAVE_ASYNC_LDS 1
#else
#define HAVE_ASYNC_LDS 0
#endif

#if HAVE_ASYNC_LDS
typedef __attribute__((address_space(1))) v4i as1_v4i;
typedef __attribute__((address_space(3))) v4i as3_v4i;
__device__ __forceinline__ as1_v4i* gcast(const void* p) {
  return (as1_v4i*)(unsigned long long)p;
}
__device__ __forceinline__ as3_v4i* lcast(void* p) {
  return (as3_v4i*)(unsigned int)(unsigned long long)p;  // low 32b = LDS offset
}
#define ASYNC_B128(src, dst) \
  __builtin_amdgcn_global_load_async_to_lds_b128(gcast(src), lcast(dst), 0, 0)
#endif

__device__ __forceinline__ void wait_async0() {
#if __has_builtin(__builtin_amdgcn_s_wait_asynccnt)
  __builtin_amdgcn_s_wait_asynccnt(0);
#else
  asm volatile("s_wait_asynccnt 0x0" ::: "memory");
#endif
}

// ---------- bf16 helpers (round-to-nearest-even) ----------
__device__ __forceinline__ bf16_t f2bf(float f) {
  unsigned u = __builtin_bit_cast(unsigned, f);
  unsigned r = (u + 0x7FFFu + ((u >> 16) & 1u)) >> 16;
  return __builtin_bit_cast(bf16_t, (unsigned short)r);
}
__device__ __forceinline__ float bf2f(bf16_t h) {
  unsigned u = ((unsigned)__builtin_bit_cast(unsigned short, h)) << 16;
  return __builtin_bit_cast(float, u);
}

// ---------- 16-lane butterfly reductions (DPP16, pure VALU) ----------
#if __has_builtin(__builtin_amdgcn_update_dpp)
template <int CTRL>
__device__ __forceinline__ float dpp_f(float x) {
  return __builtin_bit_cast(float,
      __builtin_amdgcn_update_dpp(0, __builtin_bit_cast(int, x),
                                  CTRL, 0xf, 0xf, true));
}
__device__ __forceinline__ float rowmax16(float x) {
  x = fmaxf(x, dpp_f<0xB1>(x));    // quad_perm [1,0,3,2]  : xor 1
  x = fmaxf(x, dpp_f<0x4E>(x));    // quad_perm [2,3,0,1]  : xor 2
  x = fmaxf(x, dpp_f<0x141>(x));   // row_half_mirror
  x = fmaxf(x, dpp_f<0x140>(x));   // row_mirror
  return x;
}
__device__ __forceinline__ float rowsum16(float x) {
  x += dpp_f<0xB1>(x);
  x += dpp_f<0x4E>(x);
  x += dpp_f<0x141>(x);
  x += dpp_f<0x140>(x);
  return x;
}
#else
__device__ __forceinline__ float rowmax16(float x) {
#pragma unroll
  for (int d = 1; d < 16; d <<= 1) x = fmaxf(x, __shfl_xor(x, d, 32));
  return x;
}
__device__ __forceinline__ float rowsum16(float x) {
#pragma unroll
  for (int d = 1; d < 16; d <<= 1) x += __shfl_xor(x, d, 32);
  return x;
}
#endif

// ---------- WMMA fragment loaders (per ISA 7.12.2 layouts) ----------
template <typename T>
__device__ __forceinline__ v16bf load_a_frag(const T* rowbase, int lane) {
  const int koff = (lane >> 4) * 8;
  v8bf lo = *(const v8bf*)(rowbase + koff);
  v8bf hi = *(const v8bf*)(rowbase + koff + 16);
  v16bf r;
#pragma unroll
  for (int i = 0; i < 8; ++i) { r[i] = lo[i]; r[i + 8] = hi[i]; }
  return r;
}
template <typename T>
__device__ __forceinline__ v16bf load_b_frag(const T* rowbase, int lane) {
  const int koff = (lane >> 4) * 8;
  v8bf lo = *(const v8bf*)(rowbase + koff);
  v8bf hi = *(const v8bf*)(rowbase + koff + 16);
  v16bf r;
#pragma unroll
  for (int i = 0; i < 8; ++i) { r[i] = lo[i]; r[i + 8] = hi[i]; }
  return r;
}

__device__ __forceinline__ v8f wmma_bf16(v16bf a, v16bf b, v8f c) {
  return __builtin_amdgcn_wmma_f32_16x16x32_bf16(
      false, a, false, b, (short)0, c, false, false);
}

// ---------- fp32 -> bf16 conversion ----------
__global__ void cvt_f32_bf16(const float* __restrict__ in,
                             bf16_t* __restrict__ out, int n) {
  int i = (blockIdx.x * blockDim.x + threadIdx.x) * 4;
  if (i >= n) return;
  float4 f = *(const float4*)(in + i);
  ushort4 u;
  u.x = __builtin_bit_cast(unsigned short, f2bf(f.x));
  u.y = __builtin_bit_cast(unsigned short, f2bf(f.y));
  u.z = __builtin_bit_cast(unsigned short, f2bf(f.z));
  u.w = __builtin_bit_cast(unsigned short, f2bf(f.w));
  *reinterpret_cast<ushort4*>(out + i) = u;
}

// ---------- tiled GEMM:  out[m][n] = sum_k A[m][k] * W[n][k]  (y = x @ W^T) ----------
// 4 waves/block, 64x64 wave tile (16 WMMA : 16 ds_load_b128 per K-step),
// double-buffered LDS, async global->LDS staging when available.
// MODE 0: bf16 out | MODE 1: bf16 sigmoid out | MODE 2: fp32 out + bias
template <int MODE>
__global__ void __launch_bounds__(128) gemm_bf16_nt(
    const bf16_t* __restrict__ A, const bf16_t* __restrict__ W,
    void* __restrict__ outp, const float* __restrict__ bias,
    int M, int N, int K) {
  __shared__ __align__(16) bf16_t As[2][128][40];  // 80B padded rows
  __shared__ __align__(16) bf16_t Bs[2][128][40];

  const int tid  = threadIdx.x;
  const int lane = tid & 31;
  const int w    = tid >> 5;       // 4 waves
  const int wr   = w & 1;          // 2 wave-rows (64 M-rows each)
  const int wc   = w >> 1;         // 2 wave-cols (64 N-cols each)
  const int m0   = blockIdx.y * 128;
  const int n0   = blockIdx.x * 128;
  const int lrow = lane & 15;

#if !HAVE_ASYNC_LDS
  v8bf ra[4], rb[4];
#endif
  auto stage = [&](int k0, int buf) {
#pragma unroll
    for (int i = 0; i < 4; ++i) {
      int c  = tid + i * 128;
      int r  = c >> 2;
      int kc = (c & 3) * 8;
#if HAVE_ASYNC_LDS
      ASYNC_B128(A + (size_t)(m0 + r) * K + k0 + kc, &As[buf][r][kc]);
      ASYNC_B128(W + (size_t)(n0 + r) * K + k0 + kc, &Bs[buf][r][kc]);
#else
      ra[i] = *(const v8bf*)(A + (size_t)(m0 + r) * K + k0 + kc);
      rb[i] = *(const v8bf*)(W + (size_t)(n0 + r) * K + k0 + kc);
#endif
    }
  };
  auto commit = [&](int buf) {
#if !HAVE_ASYNC_LDS
#pragma unroll
    for (int i = 0; i < 4; ++i) {
      int c  = tid + i * 128;
      int r  = c >> 2;
      int kc = (c & 3) * 8;
      *(v8bf*)&As[buf][r][kc] = ra[i];
      *(v8bf*)&Bs[buf][r][kc] = rb[i];
    }
#endif
    (void)buf;
  };

  v8f acc[4][4];
#pragma unroll
  for (int a = 0; a < 4; ++a)
#pragma unroll
    for (int b = 0; b < 4; ++b)
#pragma unroll
      for (int v = 0; v < 8; ++v) acc[a][b][v] = 0.0f;

  stage(0, 0);
  commit(0);
#if HAVE_ASYNC_LDS
  wait_async0();
#endif
  __syncthreads();

  int cur = 0;
  for (int k0 = 0; k0 < K; k0 += 32) {
    const bool has_next = (k0 + 32 < K);
    if (has_next) {
      stage(k0 + 32, cur ^ 1);               // overlaps with WMMA below
      if (k0 + 64 < K) {
        __builtin_prefetch(A + (size_t)(m0 + tid) * K + k0 + 64, 0, 1);
        __builtin_prefetch(W + (size_t)(n0 + tid) * K + k0 + 64, 0, 1);
      }
    }

    v16bf af[4], bfr[4];
#pragma unroll
    for (int mt = 0; mt < 4; ++mt)
      af[mt] = load_a_frag(&As[cur][wr * 64 + mt * 16 + lrow][0], lane);
#pragma unroll
    for (int nt = 0; nt < 4; ++nt)
      bfr[nt] = load_b_frag(&Bs[cur][wc * 64 + nt * 16 + lrow][0], lane);
#pragma unroll
    for (int mt = 0; mt < 4; ++mt)
#pragma unroll
      for (int nt = 0; nt < 4; ++nt)
        acc[mt][nt] = wmma_bf16(af[mt], bfr[nt], acc[mt][nt]);

    if (has_next) {
      commit(cur ^ 1);
#if HAVE_ASYNC_LDS
      wait_async0();
#endif
    }
    __syncthreads();
    cur ^= 1;
  }

  // epilogue (C layout: row = v + 8*(lane>=16), col = lane&15)
  const int n  = lane & 15;
  const int mb = (lane >> 4) * 8;
#pragma unroll
  for (int mt = 0; mt < 4; ++mt)
#pragma unroll
    for (int nt = 0; nt < 4; ++nt)
#pragma unroll
      for (int v = 0; v < 8; ++v) {
        int m  = m0 + wr * 64 + mt * 16 + mb + v;
        int nc = n0 + wc * 64 + nt * 16 + n;
        float val = acc[mt][nt][v];
        if (MODE == 0) {
          ((bf16_t*)outp)[(size_t)m * N + nc] = f2bf(val);
        } else if (MODE == 1) {
          float s = 1.0f / (1.0f + __expf(-val));
          ((bf16_t*)outp)[(size_t)m * N + nc] = f2bf(s);
        } else {
          ((float*)outp)[(size_t)m * N + nc] = val + bias[nc];
        }
      }
}

// ---------- fused causal flash-attention + sigmoid gate ----------
// 1 workgroup = (b, h, 64-query block); 4 waves x 16 query rows each.
// Double-buffered K/V staging (K async when available), DPP softmax.
__global__ void __launch_bounds__(128) attn_fused(
    const bf16_t* __restrict__ Q, const bf16_t* __restrict__ Kk,
    const bf16_t* __restrict__ V, const bf16_t* __restrict__ G,
    bf16_t* __restrict__ C) {
  __shared__ __align__(16) bf16_t Kt[2][32][72];   // [key][hd], 144B rows
  __shared__ __align__(16) bf16_t Vt[2][64][40];   // transposed [hd][key], 80B rows
  __shared__ __align__(16) bf16_t Pt[4][16][40];   // per-wave P buffer

  const int tid  = threadIdx.x;
  const int lane = tid & 31;
  const int wid  = tid >> 5;          // 0..3
  const int blk  = blockIdx.x;
  const int qblk = blk & 31;          // S/64 = 32
  const int h    = (blk >> 5) & 15;
  const int b    = blk >> 9;
  const int q0   = qblk * 64;

  const size_t base = ((size_t)b * SS) * DD + (size_t)h * HDD;
  const bf16_t* qbp = Q + base;
  const bf16_t* kbp = Kk + base;
  const bf16_t* vbp = V + base;
  const bf16_t* gbp = G + base;
  bf16_t*       cbp = C + base;

  const int lrow = lane & 15;
  const int n    = lrow;
  const int mb   = (lane >> 4) * 8;

  // Q A-fragments for this wave's 16 rows (hd 0..31 and 32..63)
  const int qr_frag = q0 + wid * 16 + lrow;
  v16bf qf0 = load_a_frag(qbp + (size_t)qr_frag * DD, lane);
  v16bf qf1 = load_a_frag(qbp + (size_t)qr_frag * DD + 32, lane);

  // staging coords: 2 chunks per thread (128 thr, 32x64 bf16 tiles)
  const int c0r = tid >> 3,         c0k = (tid & 7) * 8;
  const int c1r = (tid + 128) >> 3, c1k = ((tid + 128) & 7) * 8;
#if !HAVE_ASYNC_LDS
  v8bf kreg[2];
#endif
  v8bf vreg[2];
  auto k_stage = [&](int kb0, int buf) {
#if HAVE_ASYNC_LDS
    ASYNC_B128(kbp + (size_t)(kb0 + c0r) * DD + c0k, &Kt[buf][c0r][c0k]);
    ASYNC_B128(kbp + (size_t)(kb0 + c1r) * DD + c1k, &Kt[buf][c1r][c1k]);
#else
    kreg[0] = *(const v8bf*)(kbp + (size_t)(kb0 + c0r) * DD + c0k);
    kreg[1] = *(const v8bf*)(kbp + (size_t)(kb0 + c1r) * DD + c1k);
#endif
  };
  auto k_commit = [&](int buf) {
#if !HAVE_ASYNC_LDS
    *(v8bf*)&Kt[buf][c0r][c0k] = kreg[0];
    *(v8bf*)&Kt[buf][c1r][c1k] = kreg[1];
#endif
    (void)buf;
  };
  auto v_load = [&](int kb0) {
    vreg[0] = *(const v8bf*)(vbp + (size_t)(kb0 + c0r) * DD + c0k);
    vreg[1] = *(const v8bf*)(vbp + (size_t)(kb0 + c1r) * DD + c1k);
  };
  auto v_store = [&](int buf) {
#pragma unroll
    for (int j = 0; j < 8; ++j) {        // V transpose scatter
      Vt[buf][c0k + j][c0r] = vreg[0][j];
      Vt[buf][c1k + j][c1r] = vreg[1][j];
    }
  };

  v8f acc[4];
  float mr[8], lr[8];
#pragma unroll
  for (int t = 0; t < 4; ++t)
#pragma unroll
    for (int v = 0; v < 8; ++v) acc[t][v] = 0.0f;
#pragma unroll
  for (int v = 0; v < 8; ++v) { mr[v] = -3.0e38f; lr[v] = 0.0f; }

  const int kend = q0 + 64;           // causal upper bound for this block
  k_stage(0, 0);
  v_load(0);
  k_commit(0);
  v_store(0);
#if HAVE_ASYNC_LDS
  wait_async0();
#endif
  __syncthreads();

  int cur = 0;
  for (int kb0 = 0; kb0 < kend; kb0 += 32) {
    const bool has_next = (kb0 + 32 < kend);
    if (has_next) {
      k_stage(kb0 + 32, cur ^ 1);     // overlaps with math below
      v_load(kb0 + 32);
    }

    // S = Q @ K^T : two 16x16 score tiles, contraction over hd
    v8f s[2];
#pragma unroll
    for (int nt = 0; nt < 2; ++nt)
#pragma unroll
      for (int v = 0; v < 8; ++v) s[nt][v] = 0.0f;
#pragma unroll
    for (int nt = 0; nt < 2; ++nt) {
      v16bf kf0 = load_b_frag(&Kt[cur][nt * 16 + lrow][0], lane);
      v16bf kf1 = load_b_frag(&Kt[cur][nt * 16 + lrow][32], lane);
      s[nt] = wmma_bf16(qf0, kf0, s[nt]);
      s[nt] = wmma_bf16(qf1, kf1, s[nt]);
    }

    // scale (1/sqrt(64)), causal mask, online softmax (DPP reductions)
#pragma unroll
    for (int v = 0; v < 8; ++v) {
      const int qr = q0 + wid * 16 + mb + v;
#pragma unroll
      for (int nt = 0; nt < 2; ++nt) {
        int key = kb0 + nt * 16 + n;
        float x = s[nt][v] * 0.125f;
        s[nt][v] = (key > qr) ? -3.0e38f : x;
      }
      float rm   = rowmax16(fmaxf(s[0][v], s[1][v]));
      float nm   = fmaxf(mr[v], rm);
      float corr = __expf(mr[v] - nm);
      float p0   = __expf(s[0][v] - nm);
      float p1   = __expf(s[1][v] - nm);
      float rs   = rowsum16(p0 + p1);
      lr[v] = lr[v] * corr + rs;
      mr[v] = nm;
#pragma unroll
      for (int t = 0; t < 4; ++t) acc[t][v] *= corr;
      Pt[wid][mb + v][n]      = f2bf(p0);
      Pt[wid][mb + v][16 + n] = f2bf(p1);
    }

    // ctx += P @ V : contraction over 32 keys, 4 hd tiles
    v16bf pf = load_a_frag(&Pt[wid][lrow][0], lane);
#pragma unroll
    for (int t = 0; t < 4; ++t) {
      v16bf vf = load_b_frag(&Vt[cur][t * 16 + lrow][0], lane);
      acc[t] = wmma_bf16(pf, vf, acc[t]);
    }

    if (has_next) {
      k_commit(cur ^ 1);
      v_store(cur ^ 1);
#if HAVE_ASYNC_LDS
      wait_async0();
#endif
    }
    __syncthreads();
    cur ^= 1;
  }

  // normalize, apply sigmoid gate, emit bf16 ctx
#pragma unroll
  for (int v = 0; v < 8; ++v) {
    float inv = 1.0f / lr[v];
    int r = q0 + wid * 16 + mb + v;
#pragma unroll
    for (int t = 0; t < 4; ++t) {
      int col = t * 16 + n;
      float cv = acc[t][v] * inv * bf2f(gbp[(size_t)r * DD + col]);
      cbp[(size_t)r * DD + col] = f2bf(cv);
    }
  }
}

// ---------- host launch ----------
extern "C" void kernel_launch(void* const* d_in, const int* in_sizes, int n_in,
                              void* d_out, int out_size, void* d_ws, size_t ws_size,
                              hipStream_t stream) {
  const float* x  = (const float*)d_in[0];
  const float* Wq = (const float*)d_in[1];
  const float* Wk = (const float*)d_in[2];
  const float* Wv = (const float*)d_in[3];
  const float* Wg = (const float*)d_in[4];
  const float* Wo = (const float*)d_in[5];
  const float* bo = (const float*)d_in[6];

  char* ws = (char*)d_ws;
  const size_t MB = 1024ull * 1024ull;
  bf16_t* xb  = (bf16_t*)(ws + 0);        // 8 MB  [BSZ, D]
  bf16_t* wqb = (bf16_t*)(ws + 8  * MB);  // 2 MB  [D, D]
  bf16_t* wkb = (bf16_t*)(ws + 10 * MB);
  bf16_t* wvb = (bf16_t*)(ws + 12 * MB);
  bf16_t* wgb = (bf16_t*)(ws + 14 * MB);
  bf16_t* wob = (bf16_t*)(ws + 16 * MB);
  bf16_t* qb  = (bf16_t*)(ws + 18 * MB);  // 8 MB
  bf16_t* kb  = (bf16_t*)(ws + 26 * MB);  // 8 MB
  bf16_t* vb  = (bf16_t*)(ws + 34 * MB);  // 8 MB
  bf16_t* gb  = (bf16_t*)(ws + 42 * MB);  // 8 MB (sigmoid applied)
  bf16_t* cb  = (bf16_t*)(ws + 50 * MB);  // 8 MB gated ctx

  cvt_f32_bf16<<<(BSZ * DD / 4) / 256, 256, 0, stream>>>(x, xb, BSZ * DD);
  cvt_f32_bf16<<<(DD * DD / 4) / 256, 256, 0, stream>>>(Wq, wqb, DD * DD);
  cvt_f32_bf16<<<(DD * DD / 4) / 256, 256, 0, stream>>>(Wk, wkb, DD * DD);
  cvt_f32_bf16<<<(DD * DD / 4) / 256, 256, 0, stream>>>(Wv, wvb, DD * DD);
  cvt_f32_bf16<<<(DD * DD / 4) / 256, 256, 0, stream>>>(Wg, wgb, DD * DD);
  cvt_f32_bf16<<<(DD * DD / 4) / 256, 256, 0, stream>>>(Wo, wob, DD * DD);

  dim3 g(DD / 128, BSZ / 128);  // (8, 32)
  gemm_bf16_nt<0><<<g, 128, 0, stream>>>(xb, wqb, qb, nullptr, BSZ, DD, DD);
  gemm_bf16_nt<0><<<g, 128, 0, stream>>>(xb, wkb, kb, nullptr, BSZ, DD, DD);
  gemm_bf16_nt<0><<<g, 128, 0, stream>>>(xb, wvb, vb, nullptr, BSZ, DD, DD);
  gemm_bf16_nt<1><<<g, 128, 0, stream>>>(xb, wgb, gb, nullptr, BSZ, DD, DD);

  attn_fused<<<BB * HH * (SS / 64), 128, 0, stream>>>(qb, kb, vb, gb, cb);

  gemm_bf16_nt<2><<<g, 128, 0, stream>>>(cb, wob, d_out, bo, BSZ, DD, DD);
}
